// VectorizedLSTM_17489106829868
// MI455X (gfx1250) — matrix operand order repaired and checked by reading, not compiled, not run on hardware
//
#include <hip/hip_runtime.h>

// ---------------------------------------------------------------------------
// Persistent WMMA LSTM for gfx1250 (MI455X, wave32, WGP, 320KB LDS).
//   prep kernel : fp32 -> bf16 staging of x and W, zero h0, reset barrier.
//   scan kernel : 64 persistent WGs (16 hidden cols each), 256 thr = 8 waves.
//                 Per step: gates = [h, x_t](64x2048) @ Wslice^T(2048x64)
//                 via v_wmma_f32_16x16x32_bf16, B operand resident in LDS.
// ---------------------------------------------------------------------------

typedef __attribute__((ext_vector_type(16))) __bf16  v16bf;
typedef __attribute__((ext_vector_type(8)))  float    v8f;
typedef __attribute__((ext_vector_type(4)))  unsigned u32x4;

union Frag { v16bf v; u32x4 q[2]; };

#define BSZ   64
#define LSEQ  512
#define DIN   1024
#define HD    1024
#define K2    2048           // concat K = HD + DIN
#define NWG   64             // HD / 16 column groups

// workspace layout (bytes)
#define WS_CTR   0
#define WS_HBUF  256                                    // 2 * 64*1024 bf16 = 262144 B
#define WS_W16   (WS_HBUF + 2 * BSZ * HD * 2)           // 4096*2048 bf16 = 16 MiB
#define WS_X16   (WS_W16 + 4 * HD * K2 * 2)             // 64*512*1024 bf16 = 64 MiB

__device__ __forceinline__ float fsig(float x) {
  return 1.0f / (1.0f + __expf(-x));
}
__device__ __forceinline__ float ftanh(float x) {
  float e = __expf(-2.0f * x);
  return (1.0f - e) / (1.0f + e);
}

// ---------------------------------------------------------------------------
__global__ void lstm_prep(const float* __restrict__ x, const float* __restrict__ W,
                          __bf16* __restrict__ X16, __bf16* __restrict__ W16,
                          __bf16* __restrict__ hbuf, unsigned* __restrict__ counter) {
  size_t i = (size_t)blockIdx.x * blockDim.x + threadIdx.x;
  size_t stride = (size_t)gridDim.x * blockDim.x;
  if (i == 0) *counter = 0u;                       // reset device barrier each launch
  const size_t nx = (size_t)BSZ * LSEQ * DIN;      // 33,554,432
  const size_t nw = (size_t)4 * HD * K2;           //  8,388,608
  for (size_t j = i; j < nx; j += stride) X16[j] = (__bf16)x[j];
  for (size_t j = i; j < nw; j += stride) W16[j] = (__bf16)W[j];
  for (size_t j = i; j < (size_t)2 * BSZ * HD; j += stride) hbuf[j] = (__bf16)0.0f;
}

// ---------------------------------------------------------------------------
// Dynamic LDS layout (per WG):
//   [0, 256K)        : W slice, 64 rows x 2048 bf16   (row = gate*16 + col)
//   [256K, +32K)     : gate partials, 2 khalves x 4 gates x 64 x 16 fp32
//   [+4K)            : c state, 64 x 16 fp32
//   [+256)           : bias, 64 fp32
#define SM_W   0
#define SM_G   (64 * K2 * 2)
#define SM_C   (SM_G + 8 * 1024 * 4)
#define SM_B   (SM_C + 1024 * 4)
#define SM_SZ  (SM_B + 64 * 4)

__global__ __launch_bounds__(256, 1)
void lstm_scan(const __bf16* __restrict__ X16, const __bf16* __restrict__ W16,
               const float* __restrict__ bias, __bf16* __restrict__ hbuf,
               unsigned* __restrict__ counter, float* __restrict__ out) {
  extern __shared__ char smem[];
  __bf16* ldsW = (__bf16*)(smem + SM_W);
  float*  ldsG = (float*)(smem + SM_G);
  float*  ldsC = (float*)(smem + SM_C);
  float*  ldsB = (float*)(smem + SM_B);

  const int tid  = threadIdx.x;
  const int lane = tid & 31;
  const int wv   = tid >> 5;
  const int m    = wv & 3;          // M-tile (batch rows m*16..+15)
  const int kh   = wv >> 2;         // K half: 0 -> h part, 1 -> x part
  const int colbase = blockIdx.x * 16;

  // ---- one-time: stage this WG's 256KB weight slice into LDS -------------
  {
    u32x4* dst = (u32x4*)ldsW;                    // 64 rows * 256 u32x4/row
    for (int idx = tid; idx < 64 * 256; idx += 256) {
      int lr = idx >> 8;                          // local row 0..63
      int q  = idx & 255;
      int g  = lr >> 4, c = lr & 15;
      const u32x4* src =
          (const u32x4*)(W16 + (size_t)(g * HD + colbase + c) * K2) + q;
      dst[idx] = *src;
    }
  }
  if (tid < 64)
    ldsB[tid] = bias[(tid >> 4) * HD + colbase + (tid & 15)];
  for (int j = tid; j < 1024; j += 256) ldsC[j] = 0.0f;
  __syncthreads();

  const int n  = lane & 15;          // N (gate col) / M-row selector
  const int hi = (lane >> 4) & 1;    // lane half (K interleave)
  const int arow = m * 16 + n;       // batch row for A fragment

  for (int t = 0; t < LSEQ; ++t) {
    const __bf16* hprev = hbuf + (size_t)(t & 1) * (BSZ * HD);
    __bf16*       hnext = hbuf + (size_t)((t & 1) ^ 1) * (BSZ * HD);

    // A source for this wave's K half
    const __bf16* aptr = (kh == 0)
        ? hprev + (size_t)arow * HD
        : X16 + ((size_t)arow * LSEQ + t) * DIN;
    if (kh == 1 && t + 1 < LSEQ)     // pull next step's x_t toward L2/L0
      __builtin_prefetch(X16 + ((size_t)arow * LSEQ + (t + 1)) * DIN, 0, 3);

    v8f acc0 = {}, acc1 = {}, acc2 = {}, acc3 = {};

    #pragma unroll 4
    for (int ks = 0; ks < 32; ++ks) {
      const int kb = ks * 32;
      // A fragment: 16x32 bf16, lanes 0-15 K {kb..+15},
      //             lanes 16-31 K {kb+16..+31}  (matches B)
      Frag a;
      const __bf16* ap = aptr + kb + hi * 16;
      a.q[0] = *(const u32x4*)(ap);
      a.q[1] = *(const u32x4*)(ap + 8);

      // B fragments from LDS: 32x16 bf16, lane half selects K 0-15 / 16-31
      const int bk = kh * 1024 + kb + hi * 16;
      Frag b0, b1, b2, b3;
      { const u32x4* p = (const u32x4*)(ldsW + (size_t)(0 * 16 + n) * K2 + bk);
        b0.q[0] = p[0]; b0.q[1] = p[1]; }
      { const u32x4* p = (const u32x4*)(ldsW + (size_t)(1 * 16 + n) * K2 + bk);
        b1.q[0] = p[0]; b1.q[1] = p[1]; }
      { const u32x4* p = (const u32x4*)(ldsW + (size_t)(2 * 16 + n) * K2 + bk);
        b2.q[0] = p[0]; b2.q[1] = p[1]; }
      { const u32x4* p = (const u32x4*)(ldsW + (size_t)(3 * 16 + n) * K2 + bk);
        b3.q[0] = p[0]; b3.q[1] = p[1]; }

      acc0 = __builtin_amdgcn_wmma_f32_16x16x32_bf16(false, a.v, false, b0.v,
                                                     (short)0, acc0, false, false);
      acc1 = __builtin_amdgcn_wmma_f32_16x16x32_bf16(false, a.v, false, b1.v,
                                                     (short)0, acc1, false, false);
      acc2 = __builtin_amdgcn_wmma_f32_16x16x32_bf16(false, a.v, false, b2.v,
                                                     (short)0, acc2, false, false);
      acc3 = __builtin_amdgcn_wmma_f32_16x16x32_bf16(false, a.v, false, b3.v,
                                                     (short)0, acc3, false, false);
    }

    // ---- reduce partials through LDS (C layout: VGPR r -> M = r + hi*8) --
    #pragma unroll
    for (int r = 0; r < 8; ++r) {
      int grow = m * 16 + hi * 8 + r;
      ldsG[(kh * 4 + 0) * 1024 + grow * 16 + n] = acc0[r];
      ldsG[(kh * 4 + 1) * 1024 + grow * 16 + n] = acc1[r];
      ldsG[(kh * 4 + 2) * 1024 + grow * 16 + n] = acc2[r];
      ldsG[(kh * 4 + 3) * 1024 + grow * 16 + n] = acc3[r];
    }
    __syncthreads();

    // ---- elementwise LSTM cell: 1024 (batch,col) pairs over 256 threads --
    for (int j = 0; j < 4; ++j) {
      int p = tid + j * 256;               // p = row*16 + col
      int row = p >> 4, col = p & 15;
      float fpre = ldsG[0 * 1024 + p] + ldsG[4 * 1024 + p] + ldsB[col];
      float ipre = ldsG[1 * 1024 + p] + ldsG[5 * 1024 + p] + ldsB[16 + col];
      float gpre = ldsG[2 * 1024 + p] + ldsG[6 * 1024 + p] + ldsB[32 + col];
      float opre = ldsG[3 * 1024 + p] + ldsG[7 * 1024 + p] + ldsB[48 + col];
      float fv = fsig(fpre), iv = fsig(ipre);
      float gv = ftanh(gpre), ov = fsig(opre);
      float cn = fv * ldsC[p] + iv * gv;
      float hv = ov * ftanh(cn);
      ldsC[p] = cn;
      hnext[(size_t)row * HD + colbase + col] = (__bf16)hv;
      out[((size_t)row * LSEQ + t) * HD + colbase + col] = hv;
      if (t == LSEQ - 1) {
        size_t base = (size_t)BSZ * LSEQ * HD;
        out[base + (size_t)row * HD + colbase + col] = hv;                      // h_fin
        out[base + (size_t)BSZ * HD + (size_t)row * HD + colbase + col] = cn;   // c_fin
      }
    }

    // ---- device-wide step barrier (monotonic counter, reset by prep) -----
    __syncthreads();
    if (tid == 0) {
      __threadfence();                       // publish hnext
      atomicAdd(counter, 1u);
      unsigned target = (unsigned)(t + 1) * NWG;
      while (__atomic_load_n(counter, __ATOMIC_RELAXED) < target)
        __builtin_amdgcn_s_sleep(2);
      __threadfence();                       // acquire others' hnext
    }
    __syncthreads();
  }
}

// ---------------------------------------------------------------------------
extern "C" void kernel_launch(void* const* d_in, const int* in_sizes, int n_in,
                              void* d_out, int out_size, void* d_ws, size_t ws_size,
                              hipStream_t stream) {
  const float* x = (const float*)d_in[0];   // (64, 512, 1024)
  const float* W = (const float*)d_in[1];   // (4096, 2048)
  const float* b = (const float*)d_in[2];   // (4096,)
  float* out = (float*)d_out;               // outputs | h_fin | c_fin

  char* ws = (char*)d_ws;
  unsigned* counter = (unsigned*)(ws + WS_CTR);
  __bf16*   hbuf    = (__bf16*)(ws + WS_HBUF);
  __bf16*   W16     = (__bf16*)(ws + WS_W16);
  __bf16*   X16     = (__bf16*)(ws + WS_X16);

  lstm_prep<<<2048, 256, 0, stream>>>(x, W, X16, W16, hbuf, counter);
  lstm_scan<<<NWG, 256, SM_SZ, stream>>>(X16, W16, b, hbuf, counter, out);
}
